// MogLSTM_46093589021328
// MI455X (gfx1250) — matrix operand order repaired and checked by reading, not compiled, not run on hardware
//
#include <hip/hip_runtime.h>
#include <hip/hip_bf16.h>

// ---------------- CDNA5 WMMA types ----------------
typedef __attribute__((ext_vector_type(16))) __bf16    v16bf;
typedef __attribute__((ext_vector_type(8)))  float     v8f;
typedef __attribute__((ext_vector_type(4)))  unsigned  u32x4;

union Frag { v16bf v; u32x4 q[2]; };

// ---------------- problem dims (match reference) ----------------
#define S_LEN 512
#define BATCH 64
#define EDIM  1024
#define HDIM  1024
#define KCAT  2048   // E + H
#define NGATE 4096   // 4*H

#define NWG   64     // persistent workgroups
#define TPB   128    // 4 waves / WG  -> 256 waves total = 64 ntiles x 4 mtiles
#define NT    (NWG * TPB)   // 8192 threads; all phase trip counts divide this

// ---------------- helpers ----------------
__device__ __forceinline__ unsigned short f2bf(float f) {
    unsigned u = __float_as_uint(f);
    u += 0x7fffu + ((u >> 16) & 1u);          // round-to-nearest-even
    return (unsigned short)(u >> 16);
}
__device__ __forceinline__ float sigmoidf(float x) {
    return 1.0f / (1.0f + __expf(-x));
}

// device-wide generation barrier (64 WGs, all resident on MI455X)
__device__ __forceinline__ void grid_barrier(unsigned* cnt, unsigned* gen) {
    __threadfence();
    __syncthreads();
    if (threadIdx.x == 0) {
        unsigned old = __hip_atomic_load(gen, __ATOMIC_RELAXED, __HIP_MEMORY_SCOPE_AGENT);
        unsigned a   = __hip_atomic_fetch_add(cnt, 1u, __ATOMIC_ACQ_REL, __HIP_MEMORY_SCOPE_AGENT);
        if (a == (unsigned)(NWG - 1)) {
            __hip_atomic_store(cnt, 0u, __ATOMIC_RELAXED, __HIP_MEMORY_SCOPE_AGENT);
            __hip_atomic_fetch_add(gen, 1u, __ATOMIC_RELEASE, __HIP_MEMORY_SCOPE_AGENT);
        } else {
            while (__hip_atomic_load(gen, __ATOMIC_ACQUIRE, __HIP_MEMORY_SCOPE_AGENT) == old)
                __builtin_amdgcn_s_sleep(1);
        }
    }
    __syncthreads();
}

// ---- fragment loads (ISA 7.12.2 16-bit layouts; bf16 stored as ushort) ----
// A: 16x32 tile of act[64 x K] (row-major). lane<16: K 0-7 & 16-23; lane>=16: K 8-15 & 24-31.
__device__ __forceinline__ void load_fragA(Frag& a, const unsigned short* act,
                                           int K, int m, int kt, int lane) {
    int kb = kt * 32 + ((lane & 16) ? 8 : 0);
    const u32x4* p = (const u32x4*)(act + (size_t)m * K + kb);
    a.q[0] = p[0];   // halfwords kb .. kb+7
    a.q[1] = p[2];   // halfwords kb+16 .. kb+23
}
// B: 32x16 tile of W^T; W row-major [N x K] so B[k][n] = W[n][k]. lane<16: K 0-7 & 16-23; lane>=16: K 8-15 & 24-31.
__device__ __forceinline__ void load_fragB(Frag& b, const unsigned short* w,
                                           int K, int n, int kt, int lane) {
    int kb = kt * 32 + ((lane & 16) ? 8 : 0);
    const u32x4* p = (const u32x4*)(w + (size_t)n * K + kb);
    b.q[0] = p[0];
    b.q[1] = p[2];
}

// ---- mogrifier round: T[b,n] = 2*sigmoid( (act @ W^T)[b,n] ) * T[b,n], refresh bf16 shadow
__device__ __forceinline__ void mog_matmul_update(const unsigned short* actB,
                                                  const unsigned short* Wb,
                                                  float* T, unsigned short* Tb,
                                                  int mtile, int ntile, int lane) {
    v8f acc = {};
    int m = mtile * 16 + (lane & 15);
    int n = ntile * 16 + (lane & 15);
    for (int kt = 0; kt < 32; ++kt) {            // K = 1024
        Frag a, b;
        load_fragA(a, actB, HDIM, m, kt, lane);
        load_fragB(b, Wb,   HDIM, n, kt, lane);
        acc = __builtin_amdgcn_wmma_f32_16x16x32_bf16(false, a.v, false, b.v,
                                                      (short)0, acc, false, false);
    }
    int rbase = mtile * 16 + ((lane & 16) ? 8 : 0);
    int col   = ntile * 16 + (lane & 15);
#pragma unroll
    for (int r = 0; r < 8; ++r) {
        size_t idx = (size_t)(rbase + r) * HDIM + col;
        float nv = 2.0f * sigmoidf(acc[r]) * T[idx];
        T[idx]  = nv;
        Tb[idx] = f2bf(nv);
    }
}

// ---------------- the whole recurrence in one persistent kernel ----------------
__global__ __launch_bounds__(TPB)
void moglstm_persistent(const float* __restrict__ x,      // [S,B,E]
                        const float* __restrict__ W_all,  // [4H, E+H]
                        const float* __restrict__ b_all,  // [4H]
                        const float* __restrict__ Q,      // [E,H]
                        const float* __restrict__ R,      // [H,E]
                        float* __restrict__ out,          // [S*B*H + B*H + B*H]
                        unsigned char* __restrict__ ws) {
    // ---- workspace layout (~21.3 MB) ----
    unsigned* cnt = (unsigned*)ws;
    unsigned* gen = cnt + 1;
    unsigned short* Qb   = (unsigned short*)(ws + 256);       // [E x H] bf16 (row n=e, col k=h)
    unsigned short* Rb   = Qb   + (size_t)EDIM * HDIM;        // [H x E] bf16
    unsigned short* Wb   = Rb   + (size_t)HDIM * EDIM;        // [4H x 2048] bf16
    unsigned short* xtb0 = Wb   + (size_t)NGATE * KCAT;       // bf16 shadow of xt (ping)
    unsigned short* xtb1 = xtb0 + (size_t)BATCH * EDIM;       // bf16 shadow of xt (pong)
    unsigned short* htb0 = xtb1 + (size_t)BATCH * EDIM;       // bf16 shadow of ht (ping)
    unsigned short* htb1 = htb0 + (size_t)BATCH * HDIM;       // bf16 shadow of ht (pong)
    float* xt  = (float*)(htb1 + (size_t)BATCH * HDIM);       // fp32 xt  [B,E]
    float* htf = xt  + (size_t)BATCH * EDIM;                  // fp32 ht  [B,H]
    float* ctf = htf + (size_t)BATCH * HDIM;                  // fp32 ct  [B,H]

    const int tid   = blockIdx.x * TPB + threadIdx.x;         // 0..8191
    const int lane  = threadIdx.x & 31;
    const int wave  = blockIdx.x * (TPB / 32) + (threadIdx.x >> 5); // 0..255
    const int mtile = wave & 3;                               // batch-row tile (4 x 16 = 64)
    const int ntile = wave >> 2;                              // output-col tile (64 x 16 = 1024)

    // Phase 0: bf16 weight conversion + zero state + load x_0 (all counts divide NT exactly)
    for (int i = tid; i < EDIM * HDIM;  i += NT) Qb[i] = f2bf(Q[i]);
    for (int i = tid; i < HDIM * EDIM;  i += NT) Rb[i] = f2bf(R[i]);
    for (int i = tid; i < NGATE * KCAT; i += NT) Wb[i] = f2bf(W_all[i]);
    for (int i = tid; i < BATCH * HDIM; i += NT) { htf[i] = 0.f; ctf[i] = 0.f; htb0[i] = 0; }
    for (int i = tid; i < BATCH * EDIM; i += NT) { float v = x[i]; xt[i] = v; xtb0[i] = f2bf(v); }
    grid_barrier(cnt, gen);

    unsigned short* xtbC = xtb0;  unsigned short* xtbN = xtb1;   // current / next-step shadows
    unsigned short* htbC = htb0;  unsigned short* htbN = htb1;

    for (int t = 0; t < S_LEN; ++t) {
        // 5 mogrifier rounds (odd: x <- 2*sig(h@Q^T)*x ; even: h <- 2*sig(x@R^T)*h)
        mog_matmul_update(htbC, Qb, xt,  xtbC, mtile, ntile, lane); grid_barrier(cnt, gen);
        mog_matmul_update(xtbC, Rb, htf, htbC, mtile, ntile, lane); grid_barrier(cnt, gen);
        mog_matmul_update(htbC, Qb, xt,  xtbC, mtile, ntile, lane); grid_barrier(cnt, gen);
        mog_matmul_update(xtbC, Rb, htf, htbC, mtile, ntile, lane); grid_barrier(cnt, gen);
        mog_matmul_update(htbC, Qb, xt,  xtbC, mtile, ntile, lane); grid_barrier(cnt, gen);

        // gate GEMM: z = [xt|ht] @ W_all^T + b ; 4 gate tiles per wave share the A fragment.
        // Epilogue writes ht into the *next-step* shadow (htbN) so no barrier is needed
        // between the K-loop (reads htbC/xtbC) and the rewrite.
        {
            v8f a0 = {}, a1 = {}, a2 = {}, a3 = {};
            const int m = mtile * 16 + (lane & 15);
            const int n = ntile * 16 + (lane & 15);
            for (int kt = 0; kt < 32; ++kt) {                 // K 0..1023  -> xt
                Frag a, b0, b1, b2, b3;
                load_fragA(a, xtbC, EDIM, m, kt, lane);
                load_fragB(b0, Wb, KCAT, 0 * HDIM + n, kt, lane);
                load_fragB(b1, Wb, KCAT, 1 * HDIM + n, kt, lane);
                load_fragB(b2, Wb, KCAT, 2 * HDIM + n, kt, lane);
                load_fragB(b3, Wb, KCAT, 3 * HDIM + n, kt, lane);
                a0 = __builtin_amdgcn_wmma_f32_16x16x32_bf16(false, a.v, false, b0.v, (short)0, a0, false, false);
                a1 = __builtin_amdgcn_wmma_f32_16x16x32_bf16(false, a.v, false, b1.v, (short)0, a1, false, false);
                a2 = __builtin_amdgcn_wmma_f32_16x16x32_bf16(false, a.v, false, b2.v, (short)0, a2, false, false);
                a3 = __builtin_amdgcn_wmma_f32_16x16x32_bf16(false, a.v, false, b3.v, (short)0, a3, false, false);
            }
            for (int kt = 32; kt < 64; ++kt) {                // K 1024..2047 -> ht
                Frag a, b0, b1, b2, b3;
                load_fragA(a, htbC, HDIM, m, kt - 32, lane);
                load_fragB(b0, Wb, KCAT, 0 * HDIM + n, kt, lane);
                load_fragB(b1, Wb, KCAT, 1 * HDIM + n, kt, lane);
                load_fragB(b2, Wb, KCAT, 2 * HDIM + n, kt, lane);
                load_fragB(b3, Wb, KCAT, 3 * HDIM + n, kt, lane);
                a0 = __builtin_amdgcn_wmma_f32_16x16x32_bf16(false, a.v, false, b0.v, (short)0, a0, false, false);
                a1 = __builtin_amdgcn_wmma_f32_16x16x32_bf16(false, a.v, false, b1.v, (short)0, a1, false, false);
                a2 = __builtin_amdgcn_wmma_f32_16x16x32_bf16(false, a.v, false, b2.v, (short)0, a2, false, false);
                a3 = __builtin_amdgcn_wmma_f32_16x16x32_bf16(false, a.v, false, b3.v, (short)0, a3, false, false);
            }

            const int rbase = mtile * 16 + ((lane & 16) ? 8 : 0);
            const int col   = ntile * 16 + (lane & 15);
            const float bi = b_all[0 * HDIM + col];
            const float bf = b_all[1 * HDIM + col];
            const float bg = b_all[2 * HDIM + col];
            const float bo = b_all[3 * HDIM + col];
            float* out_t = out + (size_t)t * BATCH * HDIM;
#pragma unroll
            for (int r = 0; r < 8; ++r) {
                int bb = rbase + r;
                size_t idx = (size_t)bb * HDIM + col;
                float ig = sigmoidf(a0[r] + bi);
                float fg = sigmoidf(a1[r] + bf);
                float gg = tanhf(a2[r] + bg);
                float og = sigmoidf(a3[r] + bo);
                float c  = fg * ctf[idx] + ig * gg;
                ctf[idx] = c;
                float h  = og * tanhf(c);
                htf[idx] = h;
                htbN[idx] = f2bf(h);      // next-step shadow: no WAR vs. concurrent K-loops
                out_t[idx] = h;
            }
        }

        // Overlapped with the gate GEMM phase: stage x_{t+1} into fp32 xt (dead after round 5)
        // and the *next-step* bf16 shadow xtbN (disjoint from xtbC read by K-loops).
        if (t + 1 < S_LEN) {
            const float* xtin = x + (size_t)(t + 1) * BATCH * EDIM;
            for (int i = tid; i < BATCH * EDIM; i += NT) {
                float v = xtin[i]; xt[i] = v; xtbN[i] = f2bf(v);
            }
        }
        grid_barrier(cnt, gen);           // end of step: orders ht/xt staging before round 1

        // ping-pong shadows
        unsigned short* tmp;
        tmp = xtbC; xtbC = xtbN; xtbN = tmp;
        tmp = htbC; htbC = htbN; htbN = tmp;
    }

    // tail of d_out: final (ht, ct)
    float* tail = out + (size_t)S_LEN * BATCH * HDIM;
    for (int i = tid; i < BATCH * HDIM; i += NT) {
        tail[i]                = htf[i];
        tail[BATCH * HDIM + i] = ctf[i];
    }
}

extern "C" void kernel_launch(void* const* d_in, const int* in_sizes, int n_in,
                              void* d_out, int out_size, void* d_ws, size_t ws_size,
                              hipStream_t stream) {
    const float* x     = (const float*)d_in[0];
    const float* W_all = (const float*)d_in[1];
    const float* b_all = (const float*)d_in[2];
    const float* Q     = (const float*)d_in[3];
    const float* R     = (const float*)d_in[4];
    // d_in[5] = mog_iters (5) — structure hardcoded to match reference

    // reset barrier counter/generation (capturable async memset)
    hipMemsetAsync(d_ws, 0, 256, stream);
    moglstm_persistent<<<dim3(NWG), dim3(TPB), 0, stream>>>(
        x, W_all, b_all, Q, R, (float*)d_out, (unsigned char*)d_ws);
    (void)in_sizes; (void)n_in; (void)out_size; (void)ws_size;
}
